// CausalMultiHeadSelfAttention_46273977647632
// MI455X (gfx1250) — matrix operand order repaired and checked by reading, not compiled, not run on hardware
//
#include <hip/hip_runtime.h>

// ---------------------------------------------------------------------------
// CDNA5 (gfx1250) causal MHSA with RoPE, bf16 WMMA pipeline.
//   B=2, S=2048, D=1024, H=16, dh=64, theta=10000
// ---------------------------------------------------------------------------

typedef __bf16        v16bf __attribute__((ext_vector_type(16)));
typedef float         v8f   __attribute__((ext_vector_type(8)));
typedef unsigned int  uix4  __attribute__((ext_vector_type(4)));

union Frag { v16bf v; uix4 q[2]; };

#define WMMA_BF16(A, B, C) \
  __builtin_amdgcn_wmma_f32_16x16x32_bf16(false, (A), false, (B), (short)0, (C), false, false)

__device__ __forceinline__ unsigned short f2bf(float f) {
  unsigned u = __float_as_uint(f);
  return (unsigned short)((u + 0x7fffu + ((u >> 16) & 1u)) >> 16);
}

// ---------------------------------------------------------------------------
// Kernel 1: fp32 -> bf16 conversion (grid-stride)
// ---------------------------------------------------------------------------
__global__ void cvt_f32_bf16(const float* __restrict__ src,
                             unsigned short* __restrict__ dst, int n) {
  for (int i = blockIdx.x * blockDim.x + threadIdx.x; i < n;
       i += gridDim.x * blockDim.x)
    dst[i] = f2bf(src[i]);
}

// ---------------------------------------------------------------------------
// Kernel 2: fused QKV projection + RoPE.
//   out[m][n] = sum_k x[m][k] * Wqkv[n][k],  M=4096, N=3072, K=1024
//   n < 1024 -> Q (RoPE, *0.125), < 2048 -> K (RoPE), else V (transposed store)
// Block tile 128x128, 8 waves in 4(m) x 2(n) grid, each wave 2x4 WMMA tiles.
// ---------------------------------------------------------------------------
__global__ __launch_bounds__(256)
void qkv_rope_kernel(const unsigned short* __restrict__ xb,
                     const unsigned short* __restrict__ Wqkv,
                     const int* __restrict__ tp,
                     unsigned short* __restrict__ Qb,
                     unsigned short* __restrict__ Kb,
                     unsigned short* __restrict__ Vt) {
  __shared__ unsigned short As[128 * 48];  // row stride 48 ush = 96 B (16B-mult)
  __shared__ unsigned short Bs[128 * 48];
  const int tid = threadIdx.x, lane = tid & 31, w = tid >> 5;
  const int wm = w & 3, wn = w >> 2;
  const int m0 = blockIdx.y * 128, n0 = blockIdx.x * 128;
  const int lh = lane >> 4, ll = lane & 15;

  v8f acc[2][4] = {};

  const int arow = tid >> 1, ahalf = tid & 1;
  for (int k0 = 0; k0 < 1024; k0 += 32) {
    {  // cooperative stage of A (128x32) and B (128x32) bf16 tiles
      const unsigned short* s = xb + (size_t)(m0 + arow) * 1024 + k0 + ahalf * 16;
      unsigned short* d = &As[arow * 48 + ahalf * 16];
      *(uix4*)d = *(const uix4*)s;
      *(uix4*)(d + 8) = *(const uix4*)(s + 8);
      const unsigned short* s2 = Wqkv + (size_t)(n0 + arow) * 1024 + k0 + ahalf * 16;
      unsigned short* d2 = &Bs[arow * 48 + ahalf * 16];
      *(uix4*)d2 = *(const uix4*)s2;
      *(uix4*)(d2 + 8) = *(const uix4*)(s2 + 8);
      if (k0 + 32 < 1024)
        __builtin_prefetch(s + 32, 0, 1);  // global_prefetch_b8 next k-tile
    }
    __syncthreads();

    Frag af[2], bf[4];
#pragma unroll
    for (int mt = 0; mt < 2; ++mt) {  // A: lane=row, chunks K=lh*8 and +16
      const unsigned short* p = &As[(wm * 32 + mt * 16 + ll) * 48 + lh * 8];
      af[mt].q[0] = *(const uix4*)p;
      af[mt].q[1] = *(const uix4*)(p + 16);
    }
#pragma unroll
    for (int nt = 0; nt < 4; ++nt) {  // B: lane=col, chunks K=lh*8 and +16
      const unsigned short* p = &Bs[(wn * 64 + nt * 16 + ll) * 48 + lh * 8];
      bf[nt].q[0] = *(const uix4*)p;
      bf[nt].q[1] = *(const uix4*)(p + 16);
    }
#pragma unroll
    for (int mt = 0; mt < 2; ++mt)
#pragma unroll
      for (int nt = 0; nt < 4; ++nt)
        acc[mt][nt] = WMMA_BF16(af[mt].v, bf[nt].v, acc[mt][nt]);
    __syncthreads();
  }

  // Epilogue: RoPE (Q/K) + scatter stores
  const float kRopeC = 13.287712379549449f / 64.0f;  // log2(10000)/64
#pragma unroll
  for (int mt = 0; mt < 2; ++mt) {
#pragma unroll
    for (int nt = 0; nt < 4; ++nt) {
      const int ncol = n0 + wn * 64 + nt * 16 + ll;
      const int region = ncol >> 10;  // 0=Q 1=K 2=V (uniform per tile)
      const int dl = ncol & 1023;
      const int h = dl >> 6;
      const int d = dl & 63;
      const float inv = exp2f(-(float)(d & ~1) * kRopeC);
      const float sgn = (d & 1) ? 1.0f : -1.0f;
#pragma unroll
      for (int r = 0; r < 8; ++r) {
        const int mrow = m0 + wm * 32 + mt * 16 + r + lh * 8;
        const int b = mrow >> 11, s = mrow & 2047;
        float val = acc[mt][nt][r];
        if (region < 2) {
          const float other = __shfl_xor(val, 1);  // pair partner column
          const float ang = (float)tp[s] * inv;
          float sn, c;
          __sincosf(ang, &sn, &c);
          const float out = val * c + other * (sgn * sn);
          if (region == 0)  // fold 1/sqrt(dh) into Q
            Qb[((size_t)(b * 16 + h) * 2048 + s) * 64 + d] = f2bf(out * 0.125f);
          else
            Kb[((size_t)(b * 16 + h) * 2048 + s) * 64 + d] = f2bf(out);
        } else {
          Vt[((size_t)(b * 16 + h) * 64 + d) * 2048 + s] = f2bf(val);
        }
      }
    }
  }
}

// ---------------------------------------------------------------------------
// Kernel 3: flash attention (causal, online softmax).
// Grid: x = q-block of 128 rows (16), y = b*16+h (32). 8 waves x 16 q-rows.
// Key blocks of 32 so the P tile (16x32) is exactly one bf16 A-fragment.
// ---------------------------------------------------------------------------
__global__ __launch_bounds__(256)
void flash_attn_kernel(const unsigned short* __restrict__ Qb,
                       const unsigned short* __restrict__ Kb,
                       const unsigned short* __restrict__ Vt,
                       unsigned short* __restrict__ attn) {
  __shared__ unsigned short Ksh[32 * 72];      // K tile 32 keys x 64 dh
  __shared__ unsigned short Vsh[64 * 48];      // V^T tile 64 dh x 32 keys
  __shared__ unsigned short Psh[8 * 16 * 48];  // per-wave P staging
  const int tid = threadIdx.x, lane = tid & 31, w = tid >> 5;
  const int lh = lane >> 4, ll = lane & 15;
  const int q0 = blockIdx.x * 128;
  const int bh = blockIdx.y;
  const int S = 2048;

  Frag qf[2];  // Q rows live in registers across the whole key loop
  {
    const unsigned short* qp = Qb + ((size_t)bh * S + q0 + w * 16 + ll) * 64;
#pragma unroll
    for (int ks = 0; ks < 2; ++ks) {
      const int kb = ks * 32 + lh * 8;
      qf[ks].q[0] = *(const uix4*)(qp + kb);
      qf[ks].q[1] = *(const uix4*)(qp + kb + 16);
    }
  }

  float mrow[8], lrow[8];
#pragma unroll
  for (int r = 0; r < 8; ++r) { mrow[r] = -1e30f; lrow[r] = 0.f; }
  v8f O[4] = {};

  const int nkb = (q0 + 128) >> 5;
  for (int kb = 0; kb < nkb; ++kb) {
    {  // cooperative staging of K and V^T tiles (one b128 each per thread)
      const int krow = tid >> 3, koff = (tid & 7) * 8;
      *(uix4*)&Ksh[krow * 72 + koff] =
          *(const uix4*)&Kb[((size_t)bh * S + kb * 32 + krow) * 64 + koff];
      const int vd = tid >> 2, voff = (tid & 3) * 8;
      *(uix4*)&Vsh[vd * 48 + voff] =
          *(const uix4*)&Vt[((size_t)bh * 64 + vd) * S + kb * 32 + voff];
    }
    __syncthreads();

    // scores: 16 q-rows x 32 keys = two 16x16 C tiles, K-dim = dh = 2x32
    v8f sc[2];
#pragma unroll
    for (int kt = 0; kt < 2; ++kt) {
      v8f a = {};
#pragma unroll
      for (int ks = 0; ks < 2; ++ks) {
        Frag kf;
        const unsigned short* p = &Ksh[(kt * 16 + ll) * 72 + ks * 32 + lh * 8];
        kf.q[0] = *(const uix4*)p;
        kf.q[1] = *(const uix4*)(p + 16);
        a = WMMA_BF16(qf[ks].v, kf.v, a);
      }
      sc[kt] = a;
    }

    // causal mask + online softmax (row stats per VGPR r, half-wave reduce)
    const int qrow0 = q0 + w * 16 + lh * 8;
    float p[2][8], alpha[8];
#pragma unroll
    for (int r = 0; r < 8; ++r) {
      const int qrow = qrow0 + r;
      float s0 = sc[0][r], s1 = sc[1][r];
      if (kb * 32 + ll > qrow) s0 = -1e30f;
      if (kb * 32 + 16 + ll > qrow) s1 = -1e30f;
      float mx = fmaxf(s0, s1);
#pragma unroll
      for (int m = 1; m < 16; m <<= 1) mx = fmaxf(mx, __shfl_xor(mx, m));
      const float mnew = fmaxf(mrow[r], mx);
      alpha[r] = __expf(mrow[r] - mnew);
      const float p0 = __expf(s0 - mnew), p1 = __expf(s1 - mnew);
      float rs = p0 + p1;
#pragma unroll
      for (int m = 1; m < 16; m <<= 1) rs += __shfl_xor(rs, m);
      lrow[r] = lrow[r] * alpha[r] + rs;
      mrow[r] = mnew;
      p[0][r] = p0;
      p[1][r] = p1;
    }
#pragma unroll
    for (int nt = 0; nt < 4; ++nt)
#pragma unroll
      for (int r = 0; r < 8; ++r) O[nt][r] *= alpha[r];

    // P (C-layout) -> LDS -> A-fragment layout
    unsigned short* pw = &Psh[w * 16 * 48];
#pragma unroll
    for (int kt = 0; kt < 2; ++kt)
#pragma unroll
      for (int r = 0; r < 8; ++r)
        pw[(r + lh * 8) * 48 + kt * 16 + ll] = f2bf(p[kt][r]);
    asm volatile("s_wait_dscnt 0" ::: "memory");  // drain same-wave DS stores
    Frag pf;
    {
      const unsigned short* pp = &pw[ll * 48 + lh * 8];
      pf.q[0] = *(const uix4*)pp;
      pf.q[1] = *(const uix4*)(pp + 16);
    }
#pragma unroll
    for (int nt = 0; nt < 4; ++nt) {  // O += P @ V
      Frag vf;
      const unsigned short* vp = &Vsh[(nt * 16 + ll) * 48 + lh * 8];
      vf.q[0] = *(const uix4*)vp;
      vf.q[1] = *(const uix4*)(vp + 16);
      O[nt] = WMMA_BF16(pf.v, vf.v, O[nt]);
    }
    __syncthreads();
  }

  // normalize and store as [B, S, H*dh] bf16
  const int b = bh >> 4, h = bh & 15;
#pragma unroll
  for (int r = 0; r < 8; ++r) {
    const float linv = 1.0f / lrow[r];
    const int srow = q0 + w * 16 + r + lh * 8;
    unsigned short* op = attn + ((size_t)(b * S + srow)) * 1024 + h * 64;
#pragma unroll
    for (int nt = 0; nt < 4; ++nt) op[nt * 16 + ll] = f2bf(O[nt][r] * linv);
  }
}

// ---------------------------------------------------------------------------
// Kernel 4: output projection.  out[m][n] = sum_k attn[m][k]*Wo[n][k], fp32 out
// ---------------------------------------------------------------------------
__global__ __launch_bounds__(256)
void out_proj_kernel(const unsigned short* __restrict__ attnb,
                     const unsigned short* __restrict__ Wob,
                     float* __restrict__ out) {
  __shared__ unsigned short As[128 * 48];
  __shared__ unsigned short Bs[128 * 48];
  const int tid = threadIdx.x, lane = tid & 31, w = tid >> 5;
  const int wm = w & 3, wn = w >> 2;
  const int m0 = blockIdx.y * 128, n0 = blockIdx.x * 128;
  const int lh = lane >> 4, ll = lane & 15;

  v8f acc[2][4] = {};
  const int arow = tid >> 1, ahalf = tid & 1;
  for (int k0 = 0; k0 < 1024; k0 += 32) {
    {
      const unsigned short* s = attnb + (size_t)(m0 + arow) * 1024 + k0 + ahalf * 16;
      unsigned short* d = &As[arow * 48 + ahalf * 16];
      *(uix4*)d = *(const uix4*)s;
      *(uix4*)(d + 8) = *(const uix4*)(s + 8);
      const unsigned short* s2 = Wob + (size_t)(n0 + arow) * 1024 + k0 + ahalf * 16;
      unsigned short* d2 = &Bs[arow * 48 + ahalf * 16];
      *(uix4*)d2 = *(const uix4*)s2;
      *(uix4*)(d2 + 8) = *(const uix4*)(s2 + 8);
    }
    __syncthreads();
    Frag af[2], bf[4];
#pragma unroll
    for (int mt = 0; mt < 2; ++mt) {
      const unsigned short* p = &As[(wm * 32 + mt * 16 + ll) * 48 + lh * 8];
      af[mt].q[0] = *(const uix4*)p;
      af[mt].q[1] = *(const uix4*)(p + 16);
    }
#pragma unroll
    for (int nt = 0; nt < 4; ++nt) {
      const unsigned short* p = &Bs[(wn * 64 + nt * 16 + ll) * 48 + lh * 8];
      bf[nt].q[0] = *(const uix4*)p;
      bf[nt].q[1] = *(const uix4*)(p + 16);
    }
#pragma unroll
    for (int mt = 0; mt < 2; ++mt)
#pragma unroll
      for (int nt = 0; nt < 4; ++nt)
        acc[mt][nt] = WMMA_BF16(af[mt].v, bf[nt].v, acc[mt][nt]);
    __syncthreads();
  }
#pragma unroll
  for (int mt = 0; mt < 2; ++mt)
#pragma unroll
    for (int nt = 0; nt < 4; ++nt) {
      const int ncol = n0 + wn * 64 + nt * 16 + ll;
#pragma unroll
      for (int r = 0; r < 8; ++r) {
        const int mrow = m0 + wm * 32 + mt * 16 + r + lh * 8;
        out[(size_t)mrow * 1024 + ncol] = acc[mt][nt][r];
      }
    }
}

// ---------------------------------------------------------------------------
extern "C" void kernel_launch(void* const* d_in, const int* in_sizes, int n_in,
                              void* d_out, int out_size, void* d_ws, size_t ws_size,
                              hipStream_t stream) {
  const float* x  = (const float*)d_in[0];
  const float* Wq = (const float*)d_in[1];
  const float* Wk = (const float*)d_in[2];
  const float* Wv = (const float*)d_in[3];
  const float* Wo = (const float*)d_in[4];
  const int*   tp = (const int*)d_in[5];

  const size_t MD = (size_t)4096 * 1024;  // 4M elements
  unsigned short* ws    = (unsigned short*)d_ws;
  unsigned short* xb    = ws;                    // x bf16            (4M)
  unsigned short* Wqkv  = xb + MD;               // [Wq;Wk;Wv] bf16   (3M)
  unsigned short* Wob   = Wqkv + 3 * 1024 * 1024;// Wo bf16           (1M)
  unsigned short* Qb    = Wob + 1024 * 1024;     // Q  [B,H,S,dh]     (4M)
  unsigned short* Kb    = Qb + MD;               // K  [B,H,S,dh]     (4M)
  unsigned short* Vtb   = Kb + MD;               // V^T[B,H,dh,S]     (4M)
  unsigned short* attnb = Vtb + MD;              // attn [B,S,D]      (4M)
  (void)in_sizes; (void)n_in; (void)out_size; (void)ws_size;

  cvt_f32_bf16<<<2048, 256, 0, stream>>>(x,  xb,   (int)MD);
  cvt_f32_bf16<<<1024, 256, 0, stream>>>(Wq, Wqkv,               1024 * 1024);
  cvt_f32_bf16<<<1024, 256, 0, stream>>>(Wk, Wqkv + 1024 * 1024, 1024 * 1024);
  cvt_f32_bf16<<<1024, 256, 0, stream>>>(Wv, Wqkv + 2 * 1024 * 1024, 1024 * 1024);
  cvt_f32_bf16<<<1024, 256, 0, stream>>>(Wo, Wob,  1024 * 1024);

  qkv_rope_kernel<<<dim3(24, 32), 256, 0, stream>>>(xb, Wqkv, tp, Qb, Kb, Vtb);
  flash_attn_kernel<<<dim3(16, 32), 256, 0, stream>>>(Qb, Kb, Vtb, attnb);
  out_proj_kernel<<<dim3(8, 32), 256, 0, stream>>>(attnb, Wob, (float*)d_out);
}
